// EncoderLayer_12790412607718
// MI455X (gfx1250) — matrix operand order repaired and checked by reading, not compiled, not run on hardware
//
#include <hip/hip_runtime.h>

#define BB 4
#define SS 2048
#define DDIM 512
#define HH 8
#define DH 64
#define BS (BB * SS)

typedef __bf16 bf16_t;
typedef __attribute__((ext_vector_type(8)))  __bf16 v8bf;
typedef __attribute__((ext_vector_type(16))) __bf16 v16bf;
typedef __attribute__((ext_vector_type(8)))  float  v8f;

static __device__ __forceinline__ unsigned short f2bf(float f) {
  union { float f; unsigned u; } c; c.f = f;
  unsigned u = c.u;
  u += 0x7fffu + ((u >> 16) & 1u);          // round-to-nearest-even
  return (unsigned short)(u >> 16);
}

// A-operand fragment (16x32 bf16). ISA layout: lane<16 -> M=lane, K {0..7,16..23};
// lane>=16 -> M=lane-16, K {8..15,24..31}. Two 16B loads per lane.
static __device__ __forceinline__ v16bf load_fragA(const bf16_t* p, int stride, int k0) {
  int lane = threadIdx.x & 31;
  const bf16_t* q = p + (size_t)(lane & 15) * stride + k0 + ((lane >> 4) << 3);
  union { v16bf v; v8bf h[2]; } u;
  u.h[0] = *(const v8bf*)(q);
  u.h[1] = *(const v8bf*)(q + 16);
  return u.v;
}

// B-operand fragment (32x16 bf16). Same K order as A: lane<16 -> N=lane, K {0..7,16..23};
// lane>=16 -> N=lane-16, K {8..15,24..31}.
static __device__ __forceinline__ v16bf load_fragB(const bf16_t* p, int stride, int k0) {
  int lane = threadIdx.x & 31;
  const bf16_t* q = p + (size_t)(lane & 15) * stride + k0 + ((lane >> 4) << 3);
  union { v16bf v; v8bf h[2]; } u;
  u.h[0] = *(const v8bf*)(q);
  u.h[1] = *(const v8bf*)(q + 16);
  return u.v;
}

static __device__ __forceinline__ v8f wmma_bf(v16bf a, v16bf b, v8f c) {
  return __builtin_amdgcn_wmma_f32_16x16x32_bf16(false, a, false, b, (short)0, c,
                                                 false, false);
}

static __device__ __forceinline__ v8f vzero() {
  return v8f{0.f, 0.f, 0.f, 0.f, 0.f, 0.f, 0.f, 0.f};
}

// ---------------- weight transpose + f32->bf16 convert: Wt[n][k] = W[k][n] ----
__global__ void k_transpose_cvt(const float* __restrict__ W, bf16_t* __restrict__ Wt,
                                int K, int N) {
  int n = blockIdx.x * 16 + threadIdx.x;
  int k = blockIdx.y * 16 + threadIdx.y;
  ((unsigned short*)Wt)[(size_t)n * K + k] = f2bf(W[(size_t)k * N + n]);
}

// ---------------- layernorm over D=512, one wave per row, bf16 out ------------
__global__ void k_layernorm(const float* __restrict__ X, const float* __restrict__ g,
                            const float* __restrict__ b, bf16_t* __restrict__ Y) {
  int row  = blockIdx.x * 8 + (threadIdx.x >> 5);
  int lane = threadIdx.x & 31;
  const float* x = X + (size_t)row * DDIM + lane * 16;
  float v[16];
  const float4* xv = (const float4*)x;
#pragma unroll
  for (int c = 0; c < 4; ++c) {
    float4 t = xv[c];
    v[4 * c + 0] = t.x; v[4 * c + 1] = t.y; v[4 * c + 2] = t.z; v[4 * c + 3] = t.w;
  }
  float s = 0.f, s2 = 0.f;
#pragma unroll
  for (int i = 0; i < 16; ++i) { s += v[i]; s2 += v[i] * v[i]; }
#pragma unroll
  for (int w = 1; w < 32; w <<= 1) {
    s  += __shfl_xor(s,  w, 32);
    s2 += __shfl_xor(s2, w, 32);
  }
  float mu  = s * (1.0f / DDIM);
  float var = s2 * (1.0f / DDIM) - mu * mu;
  float rs  = rsqrtf(var + 1e-5f);
  union { unsigned short u[16]; uint4 q[2]; } pk;
#pragma unroll
  for (int i = 0; i < 16; ++i)
    pk.u[i] = f2bf((v[i] - mu) * rs * g[lane * 16 + i] + b[lane * 16 + i]);
  unsigned short* dst = (unsigned short*)Y + (size_t)row * DDIM + lane * 16;
  ((uint4*)dst)[0] = pk.q[0];
  ((uint4*)dst)[1] = pk.q[1];
}

// ------- fused QKV GEMM: wave = 32x64 tile, software-pipelined fragments ------
__global__ void k_qkv_gemm(const bf16_t* __restrict__ Xln,
                           const bf16_t* __restrict__ wqT, const bf16_t* __restrict__ wkT,
                           const bf16_t* __restrict__ wvT,
                           const float* __restrict__ bq, const float* __restrict__ bk,
                           const float* __restrict__ bv,
                           bf16_t* __restrict__ qb, bf16_t* __restrict__ kb,
                           bf16_t* __restrict__ vT) {
  int n0 = blockIdx.x * 64;
  int m0 = blockIdx.y * 32;
  int lane = threadIdx.x & 31;
  const bf16_t* Bt; const float* bias; int seg; int nl;
  if (n0 < 512)       { Bt = wqT; bias = bq; seg = 0; nl = n0; }
  else if (n0 < 1024) { Bt = wkT; bias = bk; seg = 1; nl = n0 - 512; }
  else                { Bt = wvT; bias = bv; seg = 2; nl = n0 - 1024; }
  v8f acc[2][4];
#pragma unroll
  for (int mt = 0; mt < 2; ++mt)
#pragma unroll
    for (int t = 0; t < 4; ++t) acc[mt][t] = vzero();

  const bf16_t* A0 = Xln + (size_t)m0 * DDIM;
  const bf16_t* A1 = A0 + (size_t)16 * DDIM;
  const bf16_t* Bb = Bt + (size_t)nl * DDIM;

  v16bf a0 = load_fragA(A0, DDIM, 0);
  v16bf a1 = load_fragA(A1, DDIM, 0);
  v16bf b0 = load_fragB(Bb + (size_t)0  * DDIM, DDIM, 0);
  v16bf b1 = load_fragB(Bb + (size_t)16 * DDIM, DDIM, 0);
  v16bf b2 = load_fragB(Bb + (size_t)32 * DDIM, DDIM, 0);
  v16bf b3 = load_fragB(Bb + (size_t)48 * DDIM, DDIM, 0);

  for (int kk = 0; kk < DDIM; kk += 32) {
    int kn = (kk + 32 < DDIM) ? kk + 32 : 0;
    // prefetch next-k fragments before issuing this k-step's WMMAs
    v16bf na0 = load_fragA(A0, DDIM, kn);
    v16bf na1 = load_fragA(A1, DDIM, kn);
    v16bf nb0 = load_fragB(Bb + (size_t)0  * DDIM, DDIM, kn);
    v16bf nb1 = load_fragB(Bb + (size_t)16 * DDIM, DDIM, kn);
    v16bf nb2 = load_fragB(Bb + (size_t)32 * DDIM, DDIM, kn);
    v16bf nb3 = load_fragB(Bb + (size_t)48 * DDIM, DDIM, kn);
    acc[0][0] = wmma_bf(a0, b0, acc[0][0]);
    acc[1][0] = wmma_bf(a1, b0, acc[1][0]);
    acc[0][1] = wmma_bf(a0, b1, acc[0][1]);
    acc[1][1] = wmma_bf(a1, b1, acc[1][1]);
    acc[0][2] = wmma_bf(a0, b2, acc[0][2]);
    acc[1][2] = wmma_bf(a1, b2, acc[1][2]);
    acc[0][3] = wmma_bf(a0, b3, acc[0][3]);
    acc[1][3] = wmma_bf(a1, b3, acc[1][3]);
    a0 = na0; a1 = na1; b0 = nb0; b1 = nb1; b2 = nb2; b3 = nb3;
  }

  int cn = lane & 15;
#pragma unroll
  for (int mt = 0; mt < 2; ++mt) {
    int rbase = m0 + mt * 16 + ((lane >> 4) << 3);
#pragma unroll
    for (int t = 0; t < 4; ++t) {
#pragma unroll
      for (int j = 0; j < 8; ++j) {
        int row = rbase + j;
        int col = nl + t * 16 + cn;
        float v = acc[mt][t][j] + bias[col];
        if (seg == 0) {
          ((unsigned short*)qb)[(size_t)row * DDIM + col] = f2bf(v * 0.125f); // 1/sqrt(DH)
        } else if (seg == 1) {
          ((unsigned short*)kb)[(size_t)row * DDIM + col] = f2bf(v);
        } else {
          int hh = col >> 6, dd = col & 63;
          int bidx = row >> 11, sidx = row & (SS - 1);
          ((unsigned short*)vT)[(((size_t)bidx * HH + hh) * DH + dd) * SS + sidx] = f2bf(v);
        }
      }
    }
  }
}

// ---------------- flash attention: one wave per (b,h,16-query tile) -----------
__global__ void k_attn(const bf16_t* __restrict__ qb, const bf16_t* __restrict__ kb,
                       const bf16_t* __restrict__ vT, const unsigned char* __restrict__ mask,
                       bf16_t* __restrict__ ob) {
  __shared__ unsigned short lds[16 * 32];
  int qt = blockIdx.x, bh = blockIdx.y;
  int b = bh >> 3, h = bh & 7;
  int lane = threadIdx.x & 31;
  int q0 = qt * 16;
  const bf16_t* qrow = qb + ((size_t)(b * SS + q0)) * DDIM + h * DH;
  v16bf aQ0 = load_fragA(qrow, DDIM, 0);
  v16bf aQ1 = load_fragA(qrow, DDIM, 32);
  v8f o[4];
#pragma unroll
  for (int t = 0; t < 4; ++t) o[t] = vzero();
  float m[8], l[8];
#pragma unroll
  for (int j = 0; j < 8; ++j) { m[j] = -1e30f; l[j] = 0.f; }
  const unsigned char* mk = mask + (size_t)b * SS;
  const bf16_t* vbase0 = vT + ((size_t)(b * HH + h)) * DH * SS;

  for (int kbk = 0; kbk < SS; kbk += 32) {
    const bf16_t* kr0 = kb + ((size_t)(b * SS + kbk)) * DDIM + h * DH;
    const bf16_t* kr1 = kr0 + (size_t)16 * DDIM;
    // issue all operand loads for this 32-key block up front:
    // K fragments (consumed by score WMMAs) and V fragments (consumed after
    // softmax — their latency hides behind the VALU/shuffle softmax work).
    v16bf bk00 = load_fragB(kr0, DDIM, 0);
    v16bf bk01 = load_fragB(kr0, DDIM, 32);
    v16bf bk10 = load_fragB(kr1, DDIM, 0);
    v16bf bk11 = load_fragB(kr1, DDIM, 32);
    v16bf bv0 = load_fragB(vbase0 + (size_t)0  * SS + kbk, SS, 0);
    v16bf bv1 = load_fragB(vbase0 + (size_t)16 * SS + kbk, SS, 0);
    v16bf bv2 = load_fragB(vbase0 + (size_t)32 * SS + kbk, SS, 0);
    v16bf bv3 = load_fragB(vbase0 + (size_t)48 * SS + kbk, SS, 0);

    v8f s0 = vzero(), s1 = vzero();
    s0 = wmma_bf(aQ0, bk00, s0);
    s0 = wmma_bf(aQ1, bk01, s0);
    s1 = wmma_bf(aQ0, bk10, s1);
    s1 = wmma_bf(aQ1, bk11, s1);

    // mask (scores already scaled by 1/sqrt(DH) via Q): NEG/8 = -1.25e8
    bool m0b = mk[kbk + (lane & 15)] != 0;
    bool m1b = mk[kbk + 16 + (lane & 15)] != 0;
#pragma unroll
    for (int j = 0; j < 8; ++j) {
      if (m0b) s0[j] = -1.25e8f;
      if (m1b) s1[j] = -1.25e8f;
    }
    float alpha[8];
#pragma unroll
    for (int j = 0; j < 8; ++j) {
      float t = fmaxf(s0[j], s1[j]);
#pragma unroll
      for (int w = 1; w < 16; w <<= 1) t = fmaxf(t, __shfl_xor(t, w, 32));
      float mn = fmaxf(m[j], t);
      alpha[j] = __expf(m[j] - mn);
      m[j] = mn;
    }
#pragma unroll
    for (int j = 0; j < 8; ++j) {
      float p0 = __expf(s0[j] - m[j]);
      float p1 = __expf(s1[j] - m[j]);
      s0[j] = p0; s1[j] = p1;
      float rs = p0 + p1;
#pragma unroll
      for (int w = 1; w < 16; w <<= 1) rs += __shfl_xor(rs, w, 32);
      l[j] = l[j] * alpha[j] + rs;
      o[0][j] *= alpha[j]; o[1][j] *= alpha[j];
      o[2][j] *= alpha[j]; o[3][j] *= alpha[j];
    }
    // D-layout -> A-layout transpose through LDS (16 rows x 32 keys, bf16)
    int rw = (lane >> 4) << 3;
    int cw = lane & 15;
#pragma unroll
    for (int j = 0; j < 8; ++j) {
      lds[(rw + j) * 32 + cw]      = f2bf(s0[j]);
      lds[(rw + j) * 32 + 16 + cw] = f2bf(s1[j]);
    }
    __syncthreads();
    v16bf aP = load_fragA((const bf16_t*)lds, 32, 0);
    o[0] = wmma_bf(aP, bv0, o[0]);
    o[1] = wmma_bf(aP, bv1, o[1]);
    o[2] = wmma_bf(aP, bv2, o[2]);
    o[3] = wmma_bf(aP, bv3, o[3]);
    __syncthreads();
  }
  int rw = (lane >> 4) << 3;
  int cw = lane & 15;
#pragma unroll
  for (int j = 0; j < 8; ++j) {
    float inv = 1.0f / l[j];
    int row = q0 + rw + j;
#pragma unroll
    for (int t = 0; t < 4; ++t) {
      int col = h * DH + t * 16 + cw;
      ((unsigned short*)ob)[((size_t)(b * SS + row)) * DDIM + col] = f2bf(o[t][j] * inv);
    }
  }
}

// ------- generic GEMM + bias (+residual)(+relu): wave = 32x64, pipelined ------
__global__ void k_gemm_ep(const bf16_t* __restrict__ A, const bf16_t* __restrict__ Bt,
                          const float* __restrict__ bias, const float* __restrict__ resid,
                          float* __restrict__ outF, bf16_t* __restrict__ outB,
                          int N, int K, int relu) {
  int n0 = blockIdx.x * 64;
  int m0 = blockIdx.y * 32;
  int lane = threadIdx.x & 31;
  v8f acc[2][4];
#pragma unroll
  for (int mt = 0; mt < 2; ++mt)
#pragma unroll
    for (int t = 0; t < 4; ++t) acc[mt][t] = vzero();

  const bf16_t* A0 = A + (size_t)m0 * K;
  const bf16_t* A1 = A0 + (size_t)16 * K;
  const bf16_t* Bb = Bt + (size_t)n0 * K;

  v16bf a0 = load_fragA(A0, K, 0);
  v16bf a1 = load_fragA(A1, K, 0);
  v16bf b0 = load_fragB(Bb + (size_t)0  * K, K, 0);
  v16bf b1 = load_fragB(Bb + (size_t)16 * K, K, 0);
  v16bf b2 = load_fragB(Bb + (size_t)32 * K, K, 0);
  v16bf b3 = load_fragB(Bb + (size_t)48 * K, K, 0);

  for (int kk = 0; kk < K; kk += 32) {
    int kn = (kk + 32 < K) ? kk + 32 : 0;
    v16bf na0 = load_fragA(A0, K, kn);
    v16bf na1 = load_fragA(A1, K, kn);
    v16bf nb0 = load_fragB(Bb + (size_t)0  * K, K, kn);
    v16bf nb1 = load_fragB(Bb + (size_t)16 * K, K, kn);
    v16bf nb2 = load_fragB(Bb + (size_t)32 * K, K, kn);
    v16bf nb3 = load_fragB(Bb + (size_t)48 * K, K, kn);
    acc[0][0] = wmma_bf(a0, b0, acc[0][0]);
    acc[1][0] = wmma_bf(a1, b0, acc[1][0]);
    acc[0][1] = wmma_bf(a0, b1, acc[0][1]);
    acc[1][1] = wmma_bf(a1, b1, acc[1][1]);
    acc[0][2] = wmma_bf(a0, b2, acc[0][2]);
    acc[1][2] = wmma_bf(a1, b2, acc[1][2]);
    acc[0][3] = wmma_bf(a0, b3, acc[0][3]);
    acc[1][3] = wmma_bf(a1, b3, acc[1][3]);
    a0 = na0; a1 = na1; b0 = nb0; b1 = nb1; b2 = nb2; b3 = nb3;
  }

  int cn = lane & 15;
#pragma unroll
  for (int mt = 0; mt < 2; ++mt) {
    int rbase = m0 + mt * 16 + ((lane >> 4) << 3);
#pragma unroll
    for (int t = 0; t < 4; ++t) {
#pragma unroll
      for (int j = 0; j < 8; ++j) {
        int row = rbase + j;
        int col = n0 + t * 16 + cn;
        float v = acc[mt][t][j] + bias[col];
        if (resid) v += resid[(size_t)row * N + col];
        if (relu)  v = fmaxf(v, 0.f);
        if (outB) ((unsigned short*)outB)[(size_t)row * N + col] = f2bf(v);
        else      outF[(size_t)row * N + col] = v;
      }
    }
  }
}

extern "C" void kernel_launch(void* const* d_in, const int* in_sizes, int n_in,
                              void* d_out, int out_size, void* d_ws, size_t ws_size,
                              hipStream_t stream) {
  (void)in_sizes; (void)n_in; (void)out_size; (void)ws_size;
  const float*         x    = (const float*)d_in[0];
  const unsigned char* mask = (const unsigned char*)d_in[1];
  const float* ln1g = (const float*)d_in[2];
  const float* ln1b = (const float*)d_in[3];
  const float* wq   = (const float*)d_in[4];
  const float* bq   = (const float*)d_in[5];
  const float* wk   = (const float*)d_in[6];
  const float* bk   = (const float*)d_in[7];
  const float* wv   = (const float*)d_in[8];
  const float* bv   = (const float*)d_in[9];
  const float* wo   = (const float*)d_in[10];
  const float* bo   = (const float*)d_in[11];
  const float* ln2g = (const float*)d_in[12];
  const float* ln2b = (const float*)d_in[13];
  const float* w1   = (const float*)d_in[14];
  const float* b1   = (const float*)d_in[15];
  const float* w2   = (const float*)d_in[16];
  const float* b2   = (const float*)d_in[17];

  char* ws = (char*)d_ws;
  size_t off = 0;
  auto take = [&](size_t bytes) { char* p = ws + off; off += bytes; return p; };
  bf16_t* x_ln  = (bf16_t*)take((size_t)BS * DDIM * 2);      // 8 MB
  bf16_t* wqT   = (bf16_t*)take((size_t)DDIM * DDIM * 2);
  bf16_t* wkT   = (bf16_t*)take((size_t)DDIM * DDIM * 2);
  bf16_t* wvT   = (bf16_t*)take((size_t)DDIM * DDIM * 2);
  bf16_t* woT   = (bf16_t*)take((size_t)DDIM * DDIM * 2);
  bf16_t* w1T   = (bf16_t*)take((size_t)DDIM * 4 * DDIM * 2);
  bf16_t* w2T   = (bf16_t*)take((size_t)DDIM * 4 * DDIM * 2);
  bf16_t* qbuf  = (bf16_t*)take((size_t)BS * DDIM * 2);
  bf16_t* kbuf  = (bf16_t*)take((size_t)BS * DDIM * 2);
  bf16_t* vTb   = (bf16_t*)take((size_t)BS * DDIM * 2);
  bf16_t* obuf  = (bf16_t*)take((size_t)BS * DDIM * 2);
  float*  x1    = (float*) take((size_t)BS * DDIM * 4);      // 16 MB
  bf16_t* xln2  = (bf16_t*)take((size_t)BS * DDIM * 2);
  bf16_t* hbuf  = (bf16_t*)take((size_t)BS * 4 * DDIM * 2);  // 33 MB

  dim3 tb(16, 16);
  k_transpose_cvt<<<dim3(DDIM / 16, DDIM / 16), tb, 0, stream>>>(wq, wqT, DDIM, DDIM);
  k_transpose_cvt<<<dim3(DDIM / 16, DDIM / 16), tb, 0, stream>>>(wk, wkT, DDIM, DDIM);
  k_transpose_cvt<<<dim3(DDIM / 16, DDIM / 16), tb, 0, stream>>>(wv, wvT, DDIM, DDIM);
  k_transpose_cvt<<<dim3(DDIM / 16, DDIM / 16), tb, 0, stream>>>(wo, woT, DDIM, DDIM);
  k_transpose_cvt<<<dim3(4 * DDIM / 16, DDIM / 16), tb, 0, stream>>>(w1, w1T, DDIM, 4 * DDIM);
  k_transpose_cvt<<<dim3(DDIM / 16, 4 * DDIM / 16), tb, 0, stream>>>(w2, w2T, 4 * DDIM, DDIM);

  k_layernorm<<<BS / 8, 256, 0, stream>>>(x, ln1g, ln1b, x_ln);

  k_qkv_gemm<<<dim3(1536 / 64, BS / 32), 32, 0, stream>>>(
      x_ln, wqT, wkT, wvT, bq, bk, bv, qbuf, kbuf, vTb);

  k_attn<<<dim3(SS / 16, BB * HH), 32, 0, stream>>>(qbuf, kbuf, vTb, mask, obuf);

  // x1 = x + o @ wo + bo
  k_gemm_ep<<<dim3(DDIM / 64, BS / 32), 32, 0, stream>>>(
      obuf, woT, bo, x, x1, nullptr, DDIM, DDIM, 0);

  k_layernorm<<<BS / 8, 256, 0, stream>>>(x1, ln2g, ln2b, xln2);

  // h = relu(xln2 @ w1 + b1)
  k_gemm_ep<<<dim3(4 * DDIM / 64, BS / 32), 32, 0, stream>>>(
      xln2, w1T, b1, nullptr, nullptr, hbuf, 4 * DDIM, DDIM, 1);

  // out = x1 + h @ w2 + b2
  k_gemm_ep<<<dim3(DDIM / 64, BS / 32), 32, 0, stream>>>(
      hbuf, w2T, b2, x1, (float*)d_out, nullptr, DDIM, 4 * DDIM, 0);
}
